// Attention_21715354649378
// MI455X (gfx1250) — compile-verified, matrix operand order checked by reading, not run on hardware
//
#include <hip/hip_runtime.h>
#include <hip/hip_bf16.h>

typedef __bf16 bf16;
typedef __bf16 v16bf __attribute__((ext_vector_type(16)));
typedef __bf16 v8bf  __attribute__((ext_vector_type(8)));
typedef float  v8f   __attribute__((ext_vector_type(8)));

#define HEADS 4
#define DHEAD 32
#define CIN   256
#define HID   128
#define NSP   4096
#define NB    4
#define SCALE 0.17677669529663687f  // 32^-0.5

// ---------------------------------------------------------------------------
// 16-bit WMMA fragment load (A- and B^T-fragments share the same pattern):
// lane L holds row (m = L&15); its 16 K-values are two contiguous 8-element
// runs at element offsets (L>>4)*8 + {0..7} and 16 + (L>>4)*8 + {0..7}.
// => two 16-byte loads per fragment.
// ---------------------------------------------------------------------------
static __device__ __forceinline__ v16bf load_frag(const bf16* row, int c0, int half) {
  v8bf lo = *(const v8bf*)(row + c0 + half * 8);
  v8bf hi = *(const v8bf*)(row + c0 + 16 + half * 8);
  v16bf r;
#pragma unroll
  for (int i = 0; i < 8; ++i) { r[i] = lo[i]; r[i + 8] = hi[i]; }
  return r;
}

static __device__ __forceinline__ v8f wmma_bf16(v16bf a, v16bf b, v8f c) {
  return __builtin_amdgcn_wmma_f32_16x16x32_bf16(false, a, false, b, (short)0, c,
                                                 false, false);
}

// ---------------------------------------------------------------------------
// Kernel 1: convert weights to bf16
// ---------------------------------------------------------------------------
__global__ void convert_w(const float* __restrict__ wq, const float* __restrict__ wo,
                          bf16* __restrict__ wqb, bf16* __restrict__ wob) {
  int i = blockIdx.x * blockDim.x + threadIdx.x;
  if (i < 3 * HID * CIN) wqb[i] = (bf16)wq[i];
  if (i < CIN * HID)     wob[i] = (bf16)wo[i];
}

// ---------------------------------------------------------------------------
// Kernel 2: x[b,c,n] f32 -> xT[b,n,c] bf16 (LDS-tiled 32x32 transpose)
// ---------------------------------------------------------------------------
__global__ __launch_bounds__(256) void transpose_x(const float* __restrict__ x,
                                                   bf16* __restrict__ xT) {
  __shared__ float tile[32][33];
  const int tx = threadIdx.x, ty = threadIdx.y;
  const int n0 = blockIdx.x * 32, c0 = blockIdx.y * 32, b = blockIdx.z;
#pragma unroll
  for (int k = 0; k < 4; ++k) {
    int c = c0 + ty + k * 8;
    tile[ty + k * 8][tx] = x[((size_t)b * CIN + c) * NSP + n0 + tx];
  }
  __syncthreads();
#pragma unroll
  for (int k = 0; k < 4; ++k) {
    int n = n0 + ty + k * 8;
    xT[((size_t)b * NSP + n) * CIN + c0 + tx] = (bf16)tile[tx][ty + k * 8];
  }
}

// ---------------------------------------------------------------------------
// Kernel 3: QKV projection, K=256. Each wave computes two 16x16 tiles along n
// sharing the A (weight) fragment -> 16 wmma per wave, A loaded once.
// Q,K stored transposed [b,h,n,32] (SCALE folded into Q); V stored [b,h,32,n].
// ---------------------------------------------------------------------------
__global__ __launch_bounds__(128) void gemm_qkv(const bf16* __restrict__ w,
                                                const bf16* __restrict__ xT,
                                                bf16* __restrict__ qT,
                                                bf16* __restrict__ kT,
                                                bf16* __restrict__ vN) {
  const int lane = threadIdx.x & 31;
  const int wave = threadIdx.x >> 5;
  const int m = lane & 15, half = lane >> 4;
  const int n0 = (blockIdx.x * 4 + wave) * 32;  // two 16-wide n tiles
  const int o0 = blockIdx.y * 16;
  const int b = blockIdx.z;
  const bf16* arow  = w + (size_t)(o0 + m) * CIN;
  const bf16* brow0 = xT + ((size_t)b * NSP + n0 + m) * CIN;
  const bf16* brow1 = xT + ((size_t)b * NSP + n0 + 16 + m) * CIN;
  v8f acc0 = {}, acc1 = {};
#pragma unroll
  for (int c0 = 0; c0 < CIN; c0 += 32) {
    v16bf a  = load_frag(arow, c0, half);
    v16bf b0 = load_frag(brow0, c0, half);
    v16bf b1 = load_frag(brow1, c0, half);
    acc0 = wmma_bf16(a, b0, acc0);
    acc1 = wmma_bf16(a, b1, acc1);
  }
  const int which = o0 >> 7;    // 0=q 1=k 2=v
  const int ol = o0 & 127;
  const int h = ol >> 5;
  const int d0 = ol & 31;       // 0 or 16
#pragma unroll
  for (int nt = 0; nt < 2; ++nt) {
    v8f acc = nt == 0 ? acc0 : acc1;
    const int n = n0 + nt * 16 + m;
    if (which < 2) {
      // D-layout: lane's 8 values are rows M = half*8 + r -> contiguous d.
      const float sc = (which == 0) ? SCALE : 1.0f;
      v8bf pk;
#pragma unroll
      for (int r = 0; r < 8; ++r) pk[r] = (bf16)(acc[r] * sc);
      bf16* dst = (which == 0 ? qT : kT) +
                  (((size_t)b * HEADS + h) * NSP + n) * DHEAD + d0 + half * 8;
      *(v8bf*)dst = pk;
    } else {
#pragma unroll
      for (int r = 0; r < 8; ++r) {
        int d = d0 + half * 8 + r;
        vN[(((size_t)b * HEADS + h) * DHEAD + d) * NSP + n] = (bf16)acc[r];
      }
    }
  }
}

// ---------------------------------------------------------------------------
// Kernel 4: flash attention. 1 wave = 2 i-tiles (32 rows), j-loop step 64
// (4 j-tiles per softmax update to amortize shuffle/exp VALU work).
// S = Q.K^T via wmma (K-dim = d = 32 exactly); online softmax with shfl_xor
// row reductions; P relaid D->A through per-wave LDS; O += P.V via wmma.
// Output written as A2^T [b, n, h*32+d] bf16 (B-fragment layout for gemm_out).
// ---------------------------------------------------------------------------
__global__ __launch_bounds__(128) void attn_flash(const bf16* __restrict__ qT,
                                                  const bf16* __restrict__ kT,
                                                  const bf16* __restrict__ vN,
                                                  bf16* __restrict__ a2T) {
  __shared__ bf16 plds[4 * 2 * 16 * 64];  // 16 KB: [wave][i-tile][16 rows][64 cols]
  const int tid = threadIdx.x;
  const int wave = tid >> 5;
  const int lane = tid & 31;
  const int m = lane & 15, half = lane >> 4;
  const int b = blockIdx.z, h = blockIdx.y;
  const size_t bh = (size_t)b * HEADS + h;
  const bf16* Q = qT + bh * (size_t)NSP * DHEAD;
  const bf16* K = kT + bh * (size_t)NSP * DHEAD;
  const bf16* V = vN + bh * (size_t)DHEAD * NSP;
  const int ibase = blockIdx.x * 128 + wave * 32;

  v16bf qa[2];
#pragma unroll
  for (int it = 0; it < 2; ++it)
    qa[it] = load_frag(Q + (size_t)(ibase + it * 16 + m) * DHEAD, 0, half);

  v8f oacc[2][2];
  float mrow[2][8], lrow[2][8];
#pragma unroll
  for (int it = 0; it < 2; ++it) {
    v8f z = {};
    oacc[it][0] = z; oacc[it][1] = z;
#pragma unroll
    for (int r = 0; r < 8; ++r) { mrow[it][r] = -1e30f; lrow[it][r] = 0.0f; }
  }

  for (int j0 = 0; j0 < NSP; j0 += 64) {
    v16bf kb[4], vb[2][2];
#pragma unroll
    for (int t = 0; t < 4; ++t)
      kb[t] = load_frag(K + (size_t)(j0 + t * 16 + m) * DHEAD, 0, half);
#pragma unroll
    for (int dt = 0; dt < 2; ++dt)
#pragma unroll
      for (int jc = 0; jc < 2; ++jc)
        vb[dt][jc] = load_frag(V + (size_t)(dt * 16 + m) * NSP, j0 + jc * 32, half);
    if (j0 + 64 < NSP) {  // prefetch next step's K/V rows
#pragma unroll
      for (int t = 0; t < 4; ++t)
        __builtin_prefetch(K + (size_t)(j0 + 64 + t * 16 + m) * DHEAD, 0, 1);
      __builtin_prefetch(V + (size_t)m * NSP + j0 + 64, 0, 1);
      __builtin_prefetch(V + (size_t)(16 + m) * NSP + j0 + 64, 0, 1);
    }
#pragma unroll
    for (int it = 0; it < 2; ++it) {
      v8f s[4];
      v8f z = {};
#pragma unroll
      for (int t = 0; t < 4; ++t) s[t] = wmma_bf16(qa[it], kb[t], z);
      // online softmax: row M=half*8+r lives in VGPR r of one 16-lane group
#pragma unroll
      for (int r = 0; r < 8; ++r) {
        float mx = fmaxf(fmaxf(s[0][r], s[1][r]), fmaxf(s[2][r], s[3][r]));
        mx = fmaxf(mx, __shfl_xor(mx, 1));
        mx = fmaxf(mx, __shfl_xor(mx, 2));
        mx = fmaxf(mx, __shfl_xor(mx, 4));
        mx = fmaxf(mx, __shfl_xor(mx, 8));
        float mnew  = fmaxf(mrow[it][r], mx);
        float alpha = __expf(mrow[it][r] - mnew);
        mrow[it][r] = mnew;
        float sum = 0.0f;
#pragma unroll
        for (int t = 0; t < 4; ++t) {
          float p = __expf(s[t][r] - mnew);
          s[t][r] = p;
          sum += p;
        }
        sum += __shfl_xor(sum, 1);
        sum += __shfl_xor(sum, 2);
        sum += __shfl_xor(sum, 4);
        sum += __shfl_xor(sum, 8);
        lrow[it][r] = lrow[it][r] * alpha + sum;
        oacc[it][0][r] = oacc[it][0][r] * alpha;
        oacc[it][1][r] = oacc[it][1][r] * alpha;
      }
      // P: D-layout -> A-layout via LDS (per-wave region, in-wave only)
      bf16* pl = plds + (size_t)(wave * 2 + it) * (16 * 64);
#pragma unroll
      for (int t = 0; t < 4; ++t)
#pragma unroll
        for (int r = 0; r < 8; ++r)
          pl[(r + half * 8) * 64 + t * 16 + m] = (bf16)s[t][r];
      __asm__ volatile("s_wait_dscnt 0" ::: "memory");
      v16bf pa0 = load_frag(pl + m * 64, 0, half);   // j chunk 0 (cols 0..31)
      v16bf pa1 = load_frag(pl + m * 64, 32, half);  // j chunk 1 (cols 32..63)
      oacc[it][0] = wmma_bf16(pa0, vb[0][0], oacc[it][0]);
      oacc[it][1] = wmma_bf16(pa0, vb[1][0], oacc[it][1]);
      oacc[it][0] = wmma_bf16(pa1, vb[0][1], oacc[it][0]);
      oacc[it][1] = wmma_bf16(pa1, vb[1][1], oacc[it][1]);
    }
  }

  // epilogue: O /= l, store as A2^T[b, n, h*32+d]
#pragma unroll
  for (int it = 0; it < 2; ++it) {
#pragma unroll
    for (int r = 0; r < 8; ++r) {
      const float inv = 1.0f / lrow[it][r];
      const int i = ibase + it * 16 + r + half * 8;
#pragma unroll
      for (int t = 0; t < 2; ++t) {
        int d = t * 16 + m;
        a2T[((size_t)b * NSP + i) * HID + h * DHEAD + d] =
            (bf16)(oacc[it][t][r] * inv);
      }
    }
  }
}

// ---------------------------------------------------------------------------
// Kernel 5: out = Wout[256,128] . A2 + b_out, f32 output [b,256,4096].
// Dual n-tile per wave sharing the A (weight) fragment.
// ---------------------------------------------------------------------------
__global__ __launch_bounds__(128) void gemm_out(const bf16* __restrict__ w,
                                                const bf16* __restrict__ a2T,
                                                const float* __restrict__ bias,
                                                float* __restrict__ out) {
  const int lane = threadIdx.x & 31;
  const int wave = threadIdx.x >> 5;
  const int m = lane & 15, half = lane >> 4;
  const int n0 = (blockIdx.x * 4 + wave) * 32;
  const int o0 = blockIdx.y * 16;
  const int b = blockIdx.z;
  const bf16* arow  = w + (size_t)(o0 + m) * HID;
  const bf16* brow0 = a2T + ((size_t)b * NSP + n0 + m) * HID;
  const bf16* brow1 = a2T + ((size_t)b * NSP + n0 + 16 + m) * HID;
  v8f acc0 = {}, acc1 = {};
#pragma unroll
  for (int c0 = 0; c0 < HID; c0 += 32) {
    v16bf a  = load_frag(arow, c0, half);
    v16bf b0 = load_frag(brow0, c0, half);
    v16bf b1 = load_frag(brow1, c0, half);
    acc0 = wmma_bf16(a, b0, acc0);
    acc1 = wmma_bf16(a, b1, acc1);
  }
#pragma unroll
  for (int r = 0; r < 8; ++r) {
    const int o = o0 + r + half * 8;
    const float bs = bias[o];
    out[((size_t)b * CIN + o) * NSP + n0 + m]      = acc0[r] + bs;
    out[((size_t)b * CIN + o) * NSP + n0 + 16 + m] = acc1[r] + bs;
  }
}

// ---------------------------------------------------------------------------
// Workspace layout (all offsets 256B aligned), total ~24.3 MB
// ---------------------------------------------------------------------------
static constexpr size_t OFF_XT = 0;         // bf16 [4,4096,256]  8,388,608 B
static constexpr size_t OFF_WQ = 8388608;   // bf16 [384,256]       196,608 B
static constexpr size_t OFF_WO = 8585216;   // bf16 [256,128]        65,536 B
static constexpr size_t OFF_QT = 8650752;   // bf16 [4,4,4096,32] 4,194,304 B
static constexpr size_t OFF_KT = 12845056;  // bf16 [4,4,4096,32] 4,194,304 B
static constexpr size_t OFF_VN = 17039360;  // bf16 [4,4,32,4096] 4,194,304 B
static constexpr size_t OFF_A2 = 21233664;  // bf16 [4,4096,128]  4,194,304 B

extern "C" void kernel_launch(void* const* d_in, const int* in_sizes, int n_in,
                              void* d_out, int out_size, void* d_ws, size_t ws_size,
                              hipStream_t stream) {
  const float* x     = (const float*)d_in[0];
  const float* w_qkv = (const float*)d_in[1];
  const float* w_out = (const float*)d_in[2];
  const float* b_out = (const float*)d_in[3];
  float* out = (float*)d_out;
  char* ws = (char*)d_ws;
  bf16* xT  = (bf16*)(ws + OFF_XT);
  bf16* wqb = (bf16*)(ws + OFF_WQ);
  bf16* wob = (bf16*)(ws + OFF_WO);
  bf16* qT  = (bf16*)(ws + OFF_QT);
  bf16* kT  = (bf16*)(ws + OFF_KT);
  bf16* vN  = (bf16*)(ws + OFF_VN);
  bf16* a2T = (bf16*)(ws + OFF_A2);

  convert_w<<<(3 * HID * CIN + 255) / 256, 256, 0, stream>>>(w_qkv, w_out, wqb, wob);
  transpose_x<<<dim3(NSP / 32, CIN / 32, NB), dim3(32, 8), 0, stream>>>(x, xT);
  gemm_qkv<<<dim3(NSP / 128, 3 * HID / 16, NB), 128, 0, stream>>>(wqb, xT, qT, kT, vN);
  attn_flash<<<dim3(NSP / 128, HEADS, NB), 128, 0, stream>>>(qT, kT, vN, a2T);
  gemm_out<<<dim3(NSP / 128, CIN / 16, NB), 128, 0, stream>>>(wob, a2T, b_out, out);
}
